// selfattention_7791070675233
// MI455X (gfx1250) — matrix operand order repaired and checked by reading, not compiled, not run on hardware
//
#include <hip/hip_runtime.h>
#include <hip/hip_bf16.h>

// ---------------------------------------------------------------------------
// Self-attention (B=8, S=2048, D=1024) for gfx1250 (MI455X), bf16 WMMA path.
//   0) wt      : W -> Wt (transpose + fp32->bf16, once per weight matrix)
//   1) qkv_proj: Q/K = X @ W + b (row-major bf16); V stored TRANSPOSED (Vt)
//   2) score   : S = Q @ K^T            (bf16 WMMA, f32 scores)
//   3) softmax : masked row softmax     (P stored bf16)
//   4) pv      : O = P @ Vt^T           (bf16 WMMA, f32 out)
// 128-thread blocks (4 waves), 128x128 block tile, 64x64 wave tile:
// 16 WMMAs per wave per k-step vs 16 ds_load_b128 (1:1), double-buffered
// fragment-order LDS, b128 fills, global_prefetch of tile k+2.
// Workspace layout (needs >= 120 MB):
//   [0,32MB)    Q  bf16 [16384,1024]   (row-major)
//   [32,64MB)   K  bf16 [16384,1024]   (row-major)
//   [64,96MB)   Vt bf16 [8][1024,2048] (transposed per batch)
//   [96,112MB)  score f32 [2048,2048]  (per-batch, reused)
//                 ... also hosts Wt bf16 [3][1024,1024] (6MB) during qkv only
//   [112,120MB) P bf16 [2048,2048]     (per-batch, reused)
// ---------------------------------------------------------------------------

typedef __attribute__((ext_vector_type(16))) __bf16 v16bf;
typedef __attribute__((ext_vector_type(8)))  float  v8f;

#define TM 128
#define TN 128
#define TK 32
#define NT 128   // threads per block (4 waves)

__device__ __forceinline__ unsigned f2bf_u(float x) {
    union { float f; unsigned u; } v; v.f = x;
    return (v.u + 0x7FFFu + ((v.u >> 16) & 1u)) >> 16;          // RNE, low 16
}
__device__ __forceinline__ unsigned bfpack2(float a, float b) {
    return f2bf_u(a) | (f2bf_u(b) << 16);
}
__device__ __forceinline__ __bf16 f2bf(float x) {
    unsigned short s = (unsigned short)f2bf_u(x);
    __bf16 out; __builtin_memcpy(&out, &s, 2); return out;
}

// Fragment-order LDS tiles (per buffer):
//   A tile 128x32 -> Af[8][32][16]; 8 K-consecutive elems (c=8g..8g+7) of row r
//     land at &Af[r>>4][(g>>1)*16+(r&15)][(g&1)*8]
//   B tile 32x128 -> Bf[8][32][16]; 8 K-consecutive elems (k=8g..8g+7) of col n
//     land at &Bf[n>>4][(g>>1)*16+(n&15)][(g&1)*8]
// Per-lane fragment read: v16bf at &frag[sub][lane][0] (2x ds_load_b128).

// ---------------- 0) W transpose + convert: Wt[z][n][k] bf16 ----------------
__global__ __launch_bounds__(256) void wt_kernel(
    const float* __restrict__ Wq, const float* __restrict__ Wk,
    const float* __restrict__ Wv, __bf16* __restrict__ Wt)
{
    const int z = blockIdx.z;
    const float* __restrict__ W = (z == 0) ? Wq : (z == 1) ? Wk : Wv;
    __bf16* __restrict__ dst = Wt + (size_t)z * 1024 * 1024;

    int idx = blockIdx.x * 256 + threadIdx.x;     // 131072 threads: n x kg
    int n = idx & 1023, k = (idx >> 10) * 8;
    float w[8];
    #pragma unroll
    for (int e = 0; e < 8; ++e)
        w[e] = W[(size_t)(k + e) * 1024 + n];     // coalesced across n
    uint4 pk = { bfpack2(w[0], w[1]), bfpack2(w[2], w[3]),
                 bfpack2(w[4], w[5]), bfpack2(w[6], w[7]) };
    *(uint4*)(dst + (size_t)n * 1024 + k) = pk;
}

// ---------------- 1) fused QKV projection: [16384,1024] x [1024,1024] -------
__global__ __launch_bounds__(NT) void qkv_proj_kernel(
    const float* __restrict__ X, const __bf16* __restrict__ WtAll,
    const float* __restrict__ bq, const float* __restrict__ bk,
    const float* __restrict__ bv,
    __bf16* __restrict__ Qb, __bf16* __restrict__ Kb, __bf16* __restrict__ Vt)
{
    const int z = blockIdx.z;
    const __bf16* __restrict__ Wt   = WtAll + (size_t)z * 1024 * 1024;  // [n][k]
    const float*  __restrict__ bias = (z == 0) ? bq : (z == 1) ? bk : bv;

    const int m0 = blockIdx.y * TM;
    const int n0 = blockIdx.x * TN;

    __shared__ alignas(32) __bf16 Af[2][8][32][16];
    __shared__ alignas(32) __bf16 Bf[2][8][32][16];

    const int tid  = threadIdx.x;
    const int lane = tid & 31;
    const int wave = tid >> 5;                    // 4 waves, grid 2(M) x 2(N)
    const int smBase = (wave & 1) * 4;            // 4 A-subtiles (64 rows)
    const int snBase = (wave >> 1) * 4;           // 4 B-subtiles (64 cols)

    v8f acc[4][4] = {};

    auto fill = [&](int buf, int k0) {
        #pragma unroll
        for (int t = 0; t < 4; ++t) {             // 512 groups, 4 per thread
            int gidx = tid + t * NT;
            int r = gidx >> 2, g = gidx & 3;
            // A: X fp32, K-contiguous -> cvt pack -> b128 LDS store
            const float4* xp = (const float4*)(X + (size_t)(m0 + r) * 1024 + k0 + g * 8);
            float4 x0 = xp[0], x1 = xp[1];
            uint4 pk = { bfpack2(x0.x, x0.y), bfpack2(x0.z, x0.w),
                         bfpack2(x1.x, x1.y), bfpack2(x1.z, x1.w) };
            *(uint4*)&Af[buf][r >> 4][(g >> 1) * 16 + (r & 15)][(g & 1) * 8] = pk;
            // B: Wt bf16 [n][k], K-contiguous -> pure wide copy
            uint4 wv = *(const uint4*)(Wt + (size_t)(n0 + r) * 1024 + k0 + g * 8);
            *(uint4*)&Bf[buf][r >> 4][(g >> 1) * 16 + (r & 15)][(g & 1) * 8] = wv;
        }
    };

    fill(0, 0);
    __syncthreads();

    int buf = 0;
    for (int k0 = 0; k0 < 1024; k0 += TK, buf ^= 1) {
        v16bf a[4], b[4];
        #pragma unroll
        for (int i = 0; i < 4; ++i) a[i] = *(const v16bf*)&Af[buf][smBase + i][lane][0];
        #pragma unroll
        for (int j = 0; j < 4; ++j) b[j] = *(const v16bf*)&Bf[buf][snBase + j][lane][0];

        if (k0 + TK < 1024) fill(buf ^ 1, k0 + TK);
        if (k0 + 2 * TK < 1024) {
            __builtin_prefetch((const void*)(X + (size_t)(m0 + (tid >> 2)) * 1024 + k0 + 2 * TK + (tid & 3) * 8), 0, 1);
            __builtin_prefetch((const void*)(Wt + (size_t)(n0 + (tid >> 2)) * 1024 + k0 + 2 * TK + (tid & 3) * 8), 0, 1);
        }

        #pragma unroll
        for (int i = 0; i < 4; ++i)
            #pragma unroll
            for (int j = 0; j < 4; ++j)
                acc[i][j] = __builtin_amdgcn_wmma_f32_16x16x32_bf16(
                    false, a[i], false, b[j], (short)0, acc[i][j], false, false);
        __syncthreads();
    }

    // Unified epilogue: addr = obase[row*rmul + col*cmul]; a 128-row tile never
    // straddles a batch, so the Vt batch offset folds into obase.
    __bf16* obase; int rmul, cmul;
    if (z == 0)      { obase = Qb; rmul = 1024; cmul = 1; }
    else if (z == 1) { obase = Kb; rmul = 1024; cmul = 1; }
    else {
        int batch = m0 >> 11;
        obase = Vt + (size_t)batch * 2048 * 1024 - (size_t)batch * 2048;
        rmul = 1; cmul = 2048;                       // Vt[b][d=col][key=row&2047]
    }

    const int n_lane = lane & 15;
    const int m_off  = (lane >> 4) * 8;
    #pragma unroll
    for (int i = 0; i < 4; ++i)
        #pragma unroll
        for (int j = 0; j < 4; ++j) {
            int col = n0 + (snBase + j) * 16 + n_lane;
            float bb = bias[col];
            #pragma unroll
            for (int v = 0; v < 8; ++v) {
                int row = m0 + (smBase + i) * 16 + m_off + v;
                obase[row * rmul + col * cmul] = f2bf(acc[i][j][v] + bb);
            }
        }
}

// ---------------- 2) scores: S = Q @ K^T  (per batch) -----------------------
__global__ __launch_bounds__(NT) void score_kernel(
    const __bf16* __restrict__ Qb, const __bf16* __restrict__ Kb,
    float* __restrict__ Sc, int batch)
{
    const __bf16* __restrict__ Q = Qb + (size_t)batch * 2048 * 1024;
    const __bf16* __restrict__ K = Kb + (size_t)batch * 2048 * 1024;

    const int q0  = blockIdx.y * TM;
    const int kk0 = blockIdx.x * TN;

    __shared__ alignas(32) __bf16 Af[2][8][32][16];
    __shared__ alignas(32) __bf16 Bf[2][8][32][16];

    const int tid  = threadIdx.x;
    const int lane = tid & 31;
    const int wave = tid >> 5;
    const int smBase = (wave & 1) * 4;
    const int snBase = (wave >> 1) * 4;

    v8f acc[4][4] = {};

    auto fill = [&](int buf, int d0) {
        #pragma unroll
        for (int t = 0; t < 4; ++t) {
            int gidx = tid + t * NT;
            int r = gidx >> 2, g = gidx & 3;
            uint4 qv = *(const uint4*)(Q + (size_t)(q0 + r) * 1024 + d0 + g * 8);
            *(uint4*)&Af[buf][r >> 4][(g >> 1) * 16 + (r & 15)][(g & 1) * 8] = qv;
            uint4 kv = *(const uint4*)(K + (size_t)(kk0 + r) * 1024 + d0 + g * 8);
            *(uint4*)&Bf[buf][r >> 4][(g >> 1) * 16 + (r & 15)][(g & 1) * 8] = kv;  // n = key
        }
    };

    fill(0, 0);
    __syncthreads();

    int buf = 0;
    for (int d0 = 0; d0 < 1024; d0 += TK, buf ^= 1) {
        v16bf a[4], b[4];
        #pragma unroll
        for (int i = 0; i < 4; ++i) a[i] = *(const v16bf*)&Af[buf][smBase + i][lane][0];
        #pragma unroll
        for (int j = 0; j < 4; ++j) b[j] = *(const v16bf*)&Bf[buf][snBase + j][lane][0];

        if (d0 + TK < 1024) fill(buf ^ 1, d0 + TK);
        if (d0 + 2 * TK < 1024) {
            __builtin_prefetch((const void*)(Q + (size_t)(q0 + (tid >> 2)) * 1024 + d0 + 2 * TK + (tid & 3) * 8), 0, 1);
            __builtin_prefetch((const void*)(K + (size_t)(kk0 + (tid >> 2)) * 1024 + d0 + 2 * TK + (tid & 3) * 8), 0, 1);
        }

        #pragma unroll
        for (int i = 0; i < 4; ++i)
            #pragma unroll
            for (int j = 0; j < 4; ++j)
                acc[i][j] = __builtin_amdgcn_wmma_f32_16x16x32_bf16(
                    false, a[i], false, b[j], (short)0, acc[i][j], false, false);
        __syncthreads();
    }

    const int n_lane = lane & 15;
    const int m_off  = (lane >> 4) * 8;
    #pragma unroll
    for (int i = 0; i < 4; ++i)
        #pragma unroll
        for (int j = 0; j < 4; ++j) {
            int col = kk0 + (snBase + j) * 16 + n_lane;
            #pragma unroll
            for (int v = 0; v < 8; ++v) {
                int row = q0 + (smBase + i) * 16 + m_off + v;
                Sc[(size_t)row * 2048 + col] = acc[i][j][v];
            }
        }
}

// ---------------- 3) masked row softmax, P stored bf16 ----------------------
__global__ __launch_bounds__(256) void softmax_kernel(
    const float* __restrict__ Sc, const int* __restrict__ mask,
    __bf16* __restrict__ P, int batch)
{
    const int row = blockIdx.x;
    const float* __restrict__ s = Sc + (size_t)row * 2048;
    const int*   __restrict__ m = mask + (size_t)batch * 2048;
    const int tid = threadIdx.x;

    __shared__ float red[256];

    float mx = -__builtin_inff();
    for (int k = tid; k < 2048; k += 256)
        if (m[k]) mx = fmaxf(mx, s[k]);
    red[tid] = mx; __syncthreads();
    for (int off = 128; off > 0; off >>= 1) {
        if (tid < off) red[tid] = fmaxf(red[tid], red[tid + off]);
        __syncthreads();
    }
    mx = red[0];
    __syncthreads();

    float sum = 0.0f;
    for (int k = tid; k < 2048; k += 256)
        if (m[k]) sum += __expf(s[k] - mx);
    red[tid] = sum; __syncthreads();
    for (int off = 128; off > 0; off >>= 1) {
        if (tid < off) red[tid] += red[tid + off];
        __syncthreads();
    }
    const float inv = 1.0f / red[0];

    for (int k = tid; k < 2048; k += 256) {
        float p = m[k] ? __expf(s[k] - mx) * inv : 0.0f;
        P[(size_t)row * 2048 + k] = f2bf(p);
    }
}

// ---------------- 4) O = P @ V  (per batch; V consumed via Vt) --------------
__global__ __launch_bounds__(NT) void pv_kernel(
    const __bf16* __restrict__ P, const __bf16* __restrict__ VtAll,
    float* __restrict__ out, int batch)
{
    const __bf16* __restrict__ Vt = VtAll + (size_t)batch * 2048 * 1024;  // [1024 d][2048 key]
    float* __restrict__ O         = out   + (size_t)batch * 2048 * 1024;

    const int m0 = blockIdx.y * TM;      // query rows
    const int n0 = blockIdx.x * TN;      // d columns

    __shared__ alignas(32) __bf16 Af[2][8][32][16];
    __shared__ alignas(32) __bf16 Bf[2][8][32][16];

    const int tid  = threadIdx.x;
    const int lane = tid & 31;
    const int wave = tid >> 5;
    const int smBase = (wave & 1) * 4;
    const int snBase = (wave >> 1) * 4;

    v8f acc[4][4] = {};

    auto fill = [&](int buf, int k0) {
        #pragma unroll
        for (int t = 0; t < 4; ++t) {
            int gidx = tid + t * NT;
            int r = gidx >> 2, g = gidx & 3;
            uint4 pv = *(const uint4*)(P + (size_t)(m0 + r) * 2048 + k0 + g * 8);
            *(uint4*)&Af[buf][r >> 4][(g >> 1) * 16 + (r & 15)][(g & 1) * 8] = pv;
            uint4 vv = *(const uint4*)(Vt + (size_t)(n0 + r) * 2048 + k0 + g * 8);  // n = d
            *(uint4*)&Bf[buf][r >> 4][(g >> 1) * 16 + (r & 15)][(g & 1) * 8] = vv;
        }
    };

    fill(0, 0);
    __syncthreads();

    int buf = 0;
    for (int k0 = 0; k0 < 2048; k0 += TK, buf ^= 1) {
        v16bf a[4], b[4];
        #pragma unroll
        for (int i = 0; i < 4; ++i) a[i] = *(const v16bf*)&Af[buf][smBase + i][lane][0];
        #pragma unroll
        for (int j = 0; j < 4; ++j) b[j] = *(const v16bf*)&Bf[buf][snBase + j][lane][0];

        if (k0 + TK < 2048) fill(buf ^ 1, k0 + TK);
        if (k0 + 2 * TK < 2048) {
            __builtin_prefetch((const void*)(P + (size_t)(m0 + (tid >> 2)) * 2048 + k0 + 2 * TK + (tid & 3) * 8), 0, 1);
            __builtin_prefetch((const void*)(Vt + (size_t)(n0 + (tid >> 2)) * 2048 + k0 + 2 * TK + (tid & 3) * 8), 0, 1);
        }

        #pragma unroll
        for (int i = 0; i < 4; ++i)
            #pragma unroll
            for (int j = 0; j < 4; ++j)
                acc[i][j] = __builtin_amdgcn_wmma_f32_16x16x32_bf16(
                    false, a[i], false, b[j], (short)0, acc[i][j], false, false);
        __syncthreads();
    }

    const int n_lane = lane & 15;
    const int m_off  = (lane >> 4) * 8;
    #pragma unroll
    for (int i = 0; i < 4; ++i)
        #pragma unroll
        for (int j = 0; j < 4; ++j) {
            int col = n0 + (snBase + j) * 16 + n_lane;
            #pragma unroll
            for (int v = 0; v < 8; ++v) {
                int row = m0 + (smBase + i) * 16 + m_off + v;
                O[(size_t)row * 1024 + col] = acc[i][j][v];
            }
        }
}

// ---------------------------------------------------------------------------
extern "C" void kernel_launch(void* const* d_in, const int* in_sizes, int n_in,
                              void* d_out, int out_size, void* d_ws, size_t ws_size,
                              hipStream_t stream) {
    const float* X    = (const float*)d_in[0];
    const int*   mask = (const int*)  d_in[1];
    const float* Wq   = (const float*)d_in[2];
    const float* bq   = (const float*)d_in[3];
    const float* Wk   = (const float*)d_in[4];
    const float* bk   = (const float*)d_in[5];
    const float* Wv   = (const float*)d_in[6];
    const float* bv   = (const float*)d_in[7];
    float* out = (float*)d_out;

    char* ws = (char*)d_ws;
    __bf16* Qb = (__bf16*)(ws);
    __bf16* Kb = (__bf16*)(ws + ((size_t)32  << 20));
    __bf16* Vt = (__bf16*)(ws + ((size_t)64  << 20));
    float*  Sc = (float*) (ws + ((size_t)96  << 20));
    __bf16* Wt = (__bf16*)(ws + ((size_t)96  << 20));  // overlaps Sc; used only
                                                       // before score kernels run
    __bf16* P  = (__bf16*)(ws + ((size_t)112 << 20));

    // 0) W -> Wt (transpose + convert), once
    wt_kernel<<<dim3(512, 1, 3), dim3(256), 0, stream>>>(Wq, Wk, Wv, Wt);

    // 1) Q/K/V projections: grid.z selects which weight
    qkv_proj_kernel<<<dim3(8, 128, 3), dim3(NT), 0, stream>>>(
        X, Wt, bq, bk, bv, Qb, Kb, Vt);

    // 2-4) per batch: scores -> softmax -> P@V (stream-ordered; Sc overwrite of
    // Wt is safe because qkv completed before the first score kernel starts)
    for (int b = 0; b < 8; ++b) {
        score_kernel  <<<dim3(16, 16), dim3(NT), 0, stream>>>(Qb, Kb, Sc, b);
        softmax_kernel<<<dim3(2048),   dim3(256), 0, stream>>>(Sc, mask, P, b);
        pv_kernel     <<<dim3(8, 16),  dim3(NT), 0, stream>>>(P, Vt, out, b);
    }
}
